// LSTMModel_46986942218987
// MI455X (gfx1250) — matrix operand order repaired and checked by reading, not compiled, not run on hardware
//
#include <hip/hip_runtime.h>
#include <hip/hip_bf16.h>

// ---------------------------------------------------------------------------
// 25-layer LSTM (B=1000, S=256, E=64, H=8) for MI455X / gfx1250.
// Batch-partitioned: each workgroup owns BM=64 batch rows and runs the full
// S x L recurrence with all weights + h/c state resident in LDS.
// Gate GEMMs use v_wmma_f32_16x16x32_bf16 (bf16 inputs, fp32 accumulate).
// Layer 0 (K=96) is peeled so its 3-WMMA chain fully unrolls; inner layers
// are a single WMMA. Activations are branchless fast sigmoid/tanh (v_exp +
// v_rcp). The per-timestep embedding gather is software-pipelined into VGPRs;
// hot-loop barriers are LDS-only (s_wait_dscnt + split barrier) so the
// prefetch loads stay outstanding across the whole 25-layer compute.
// ---------------------------------------------------------------------------

typedef __attribute__((ext_vector_type(16))) __bf16   v16bf;
typedef __attribute__((ext_vector_type(8)))  float    v8f;
typedef __attribute__((ext_vector_type(8)))  unsigned v8u;

#define L_LAY 25
#define B_TOT 1000
#define S_SEQ 256
#define E_EMB 64
#define H_HID 8
#define NG    32            // 4 gates * H = gate columns
#define BM    64            // batch rows per workgroup
#define NT    256           // threads per block = 8 waves (wave32)
#define K0    96            // padded K for layer 0 (64 + 8 -> 96 = 3*32)
#define KI    32            // padded K for inner layers (8 + 8 -> 32)

#define WT0_ELEMS (NG * K0)                              // 3072 bf16
#define WTI_ELEMS (NG * KI)                              // 1024 bf16
#define WT_ELEMS  (WT0_ELEMS + (L_LAY - 1) * WTI_ELEMS)  // 27648 bf16

#define WT_BYTES   ((size_t)WT_ELEMS * 2)           // 55296
#define BIAS_BYTES ((size_t)L_LAY * NG * 4)         // 3200
#define ZBUF_BYTES ((size_t)BM * K0 * 2)            // 12288
#define HBUF_BYTES ((size_t)L_LAY * BM * H_HID * 4) // 51200
#define GBUF_BYTES ((size_t)BM * NG * 4)            // 8192
#define SMEM_BYTES (WT_BYTES + BIAS_BYTES + ZBUF_BYTES + 2 * HBUF_BYTES + GBUF_BYTES)

// LDS-only workgroup barrier: drain DScnt, then split barrier. Does NOT drain
// LOADcnt, so VGPR-destination global prefetch loads stay outstanding.
__device__ __forceinline__ void barrier_lds() {
    asm volatile("s_wait_dscnt 0x0\n\t"
                 "s_barrier_signal -1\n\t"
                 "s_barrier_wait -1" ::: "memory");
}

// ---- fast branchless activations: one v_exp_f32 + one v_rcp_f32 ----
__device__ __forceinline__ float fast_rcp(float x) {
    return __builtin_amdgcn_rcpf(x);
}
__device__ __forceinline__ float tanh_fast(float x) {
    // tanh(x) = 2 / (1 + exp(-2x)) - 1
    return fmaf(2.0f, fast_rcp(1.0f + __expf(-2.0f * x)), -1.0f);
}
__device__ __forceinline__ unsigned pack_bf16(float a, float b) {
    unsigned short ua = __builtin_bit_cast(unsigned short, (__bf16)a);
    unsigned short ub = __builtin_bit_cast(unsigned short, (__bf16)b);
    return ((unsigned)ub << 16) | (unsigned)ua;
}

// A-matrix fragment: 16x32 bf16, row M = lane&15, K pairs per ISA layout:
//   VGPR p<4 : K = 2p + (hi?8:0) ;  p>=4 : K = 16 + 2(p-4) + (hi?8:0)
__device__ __forceinline__ v16bf load_fragA(const __bf16* rowp, int hi) {
    v8u tmp;
#pragma unroll
    for (int p = 0; p < 8; ++p) {
        int kb = ((p < 4) ? (2 * p) : (16 + 2 * (p - 4))) + (hi ? 8 : 0);
        tmp[p] = *(const unsigned*)(rowp + kb);   // pair (kb, kb+1), dword aligned
    }
    return __builtin_bit_cast(v16bf, tmp);
}
// B-matrix fragment: 32x16 bf16 (KxN), col N = lane&15, K pairs same as A layout.
// Weights stored column-major [n][k] so pairs are contiguous dwords.
__device__ __forceinline__ v16bf load_fragB(const __bf16* colp, int hi) {
    v8u tmp;
#pragma unroll
    for (int p = 0; p < 8; ++p) {
        int kb = ((p < 4) ? (2 * p) : (16 + 2 * (p - 4))) + (hi ? 8 : 0);
        tmp[p] = *(const unsigned*)(colp + kb);
    }
    return __builtin_bit_cast(v16bf, tmp);
}

// Gate tile for one wave: KPAD/32 chained WMMAs, then branchless activation,
// result written to gbuf (fp32, BM x 32).
template <int KPAD>
__device__ __forceinline__ void layer_gates(const __bf16* __restrict__ zb,
                                            const __bf16* __restrict__ wl,
                                            const float*  __restrict__ bias_l,
                                            float* __restrict__ gbuf,
                                            int mT, int j, int hi, int ln,
                                            float ascale, float amul, float aoff)
{
    v8f acc;
    float bj = bias_l[j];
#pragma unroll
    for (int v = 0; v < 8; ++v) acc[v] = bj;

    const __bf16* arow = zb + (mT * 16 + ln) * KPAD;
    const __bf16* bcol = wl + j * KPAD;
#pragma unroll
    for (int kc = 0; kc < KPAD / 32; ++kc) {
        v16bf af = load_fragA(arow + kc * 32, hi);
        v16bf bf = load_fragB(bcol + kc * 32, hi);
        acc = __builtin_amdgcn_wmma_f32_16x16x32_bf16(
            false, af, false, bf, (short)0, acc, false, false);
    }
#pragma unroll
    for (int v = 0; v < 8; ++v) {
        float g = acc[v];
        float a = fmaf(amul, fast_rcp(1.0f + __expf(-ascale * g)), aoff);
        int row = mT * 16 + v + hi * 8;              // C/D layout: M = vgpr + hi*8
        gbuf[row * NG + j] = a;
    }
}

// Cell update for layer l, fused with staging of next layer's Z.
// Thread mapping: bl = tid>>2 (batch row), h0 = 2*(tid&3) (h pair).
__device__ __forceinline__ void cell_update_stage(int l, bool stage, bool zeroTail,
                                                  const float* __restrict__ gbuf,
                                                  float* __restrict__ hbuf,
                                                  float* __restrict__ cbuf,
                                                  __bf16* __restrict__ zbuf, int tid)
{
    const int bl = tid >> 2;
    const int h0 = (tid & 3) * 2;
    const float* gr = gbuf + bl * NG;
    float* hp = hbuf + (l * BM + bl) * H_HID + h0;
    float* cp = cbuf + (l * BM + bl) * H_HID + h0;
    float hn[2];
#pragma unroll
    for (int u = 0; u < 2; ++u) {
        int h = h0 + u;
        float f = gr[h];
        float i = gr[8 + h];
        float C = gr[16 + h];
        float o = gr[24 + h];
        float cNew = fmaf(cp[u], f, i * C);
        hn[u] = o * tanh_fast(cNew);
        cp[u] = cNew;
        hp[u] = hn[u];
    }
    if (stage) {
        __bf16* zr = zbuf + bl * KI;
        *(unsigned*)(zr + h0) = pack_bf16(hn[0], hn[1]);              // h_l (new)
        const float* hrec = hbuf + ((l + 1) * BM + bl) * H_HID + h0;  // h_{l+1} (old)
        *(unsigned*)(zr + 8 + h0) = pack_bf16(hrec[0], hrec[1]);
        if (zeroTail) {                                               // K pad 16..31
            *(unsigned*)(zr + 16 + h0) = 0u;
            *(unsigned*)(zr + 24 + h0) = 0u;
        }
    }
}

__global__ __launch_bounds__(NT) void lstm_wmma_kernel(
    const int*   __restrict__ x,     // (B,S)
    const float* __restrict__ h0,    // (L,B,H)
    const float* __restrict__ c0,    // (L,B,H)
    const float* __restrict__ emb,   // (V,E)
    const float* __restrict__ Wx0,   // (4,E,H)
    const float* __restrict__ Wh0,   // (4,H,H)
    const float* __restrict__ b0,    // (4,H)
    const float* __restrict__ Wx,    // (L-1,4,H,H)
    const float* __restrict__ Wh,    // (L-1,4,H,H)
    const float* __restrict__ bb,    // (L-1,4,H)
    const float* __restrict__ Why,   // (H,1)
    const float* __restrict__ by,    // (1,)
    float*       __restrict__ out)   // logits (B,) ++ h_f (L,B,H)
{
    extern __shared__ __align__(16) char smem[];
    __bf16* wT   = (__bf16*)(smem);
    float*  bias = (float*) (smem + WT_BYTES);
    __bf16* zbuf = (__bf16*)(smem + WT_BYTES + BIAS_BYTES);
    float*  hbuf = (float*) (smem + WT_BYTES + BIAS_BYTES + ZBUF_BYTES);
    float*  cbuf = (float*) (smem + WT_BYTES + BIAS_BYTES + ZBUF_BYTES + HBUF_BYTES);
    float*  gbuf = (float*) (smem + WT_BYTES + BIAS_BYTES + ZBUF_BYTES + 2 * HBUF_BYTES);

    const int tid  = threadIdx.x;
    const int lane = tid & 31;
    const int wid  = tid >> 5;
    const int hi   = (lane >> 4) & 1;
    const int ln   = lane & 15;
    const int mT   = wid >> 1;        // 0..3  (M tile)
    const int nT   = wid & 1;         // 0..1  (N tile)
    const int bBase = blockIdx.x * BM;

    // per-lane activation constants (gate is fixed per lane for whole kernel)
    const int   j      = nT * 16 + ln;          // gate column 0..31
    const int   gate   = j >> 3;                // 0=f,1=i (sigmoid) 2=C (tanh) 3=o
    const float ascale = (gate == 2) ? 2.0f : 1.0f;
    const float amul   = (gate == 2) ? 2.0f : 1.0f;
    const float aoff   = (gate == 2) ? -1.0f : 0.0f;

    // staging-thread mapping (also used by the prefetch pipeline)
    const int  bl_s    = tid >> 2;
    const int  q_s     = tid & 3;
    const int  bg_s    = bBase + bl_s;
    const bool valid_s = (bg_s < B_TOT);

    // ---- one-time: convert weights to bf16, column-major [n][Kpad], in LDS ----
    for (int idx = tid; idx < WT0_ELEMS; idx += NT) {
        int n = idx / K0, k = idx % K0;
        int g = n >> 3, h = n & 7;
        float v = 0.0f;
        if (k < E_EMB)              v = Wx0[(g * E_EMB + k) * H_HID + h];
        else if (k < E_EMB + H_HID) v = Wh0[(g * H_HID + (k - E_EMB)) * H_HID + h];
        wT[idx] = (__bf16)v;
    }
    for (int idx = tid; idx < (L_LAY - 1) * WTI_ELEMS; idx += NT) {
        int r  = idx % WTI_ELEMS;
        int l1 = idx / WTI_ELEMS;
        int n = r / KI, k = r % KI;
        int g = n >> 3, h = n & 7;
        float v = 0.0f;
        if (k < H_HID)          v = Wx[((l1 * 4 + g) * H_HID + k) * H_HID + h];
        else if (k < 2 * H_HID) v = Wh[((l1 * 4 + g) * H_HID + (k - H_HID)) * H_HID + h];
        wT[WT0_ELEMS + idx] = (__bf16)v;
    }
    for (int idx = tid; idx < L_LAY * NG; idx += NT) {
        int l = idx >> 5, jj = idx & 31;
        bias[idx] = (l == 0) ? b0[jj] : bb[(l - 1) * NG + jj];
    }
    // ---- init h/c state (fp32) in LDS ----
    for (int idx = tid; idx < L_LAY * BM * H_HID; idx += NT) {
        int l  = idx / (BM * H_HID);
        int r  = idx % (BM * H_HID);
        int bl = r >> 3, h = r & 7;
        int bg = bBase + bl;
        float hv = 0.0f, cv = 0.0f;
        if (bg < B_TOT) {
            hv = h0[((size_t)l * B_TOT + bg) * H_HID + h];
            cv = c0[((size_t)l * B_TOT + bg) * H_HID + h];
        }
        hbuf[idx] = hv;
        cbuf[idx] = cv;
    }

    // ---- prefetch pipeline: each thread holds its 16-float emb slice ----
    float pf[16];
    auto prefetch_emb = [&](int t) {
        int tc = (t < S_SEQ) ? t : (S_SEQ - 1);
        int xid = valid_s ? x[(size_t)bg_s * S_SEQ + tc] : 0;
        const float* erow = emb + (size_t)xid * E_EMB + q_s * 16;
#pragma unroll
        for (int jj = 0; jj < 16; jj += 4) {
            float4 v = *(const float4*)(erow + jj);
            pf[jj]     = v.x;
            pf[jj + 1] = v.y;
            pf[jj + 2] = v.z;
            pf[jj + 3] = v.w;
        }
    };
    prefetch_emb(0);
    __syncthreads();   // one-time full barrier after init

    // ================= time loop =================
    for (int t = 0; t < S_SEQ; ++t) {
        // ---- stage layer-0 Z = [emb(x_t) | h[0] | pad] : BM x K0 bf16 ----
        {
            __bf16* zr = zbuf + bl_s * K0;
#pragma unroll
            for (int jj = 0; jj < 16; jj += 2) {     // emb slice k = q*16 + jj
                float v0 = valid_s ? pf[jj]     : 0.0f;
                float v1 = valid_s ? pf[jj + 1] : 0.0f;
                *(unsigned*)(zr + q_s * 16 + jj) = pack_bf16(v0, v1);
            }
            if (q_s == 0) {                          // k = 64..71 : h[0]
                const float* hr = hbuf + bl_s * H_HID;
#pragma unroll
                for (int jj = 0; jj < 8; jj += 2)
                    *(unsigned*)(zr + 64 + jj) = pack_bf16(hr[jj], hr[jj + 1]);
            } else {                                 // k = 72..95 : zero pad
                __bf16* zt = zr + 72 + (q_s - 1) * 8;
#pragma unroll
                for (int jj = 0; jj < 8; jj += 2)
                    *(unsigned*)(zt + jj) = 0u;
            }
        }
        // issue next timestep's gather now; stays outstanding across the
        // LDS-only barriers below, hidden by 25 layers of compute
        prefetch_emb(t + 1);
        barrier_lds();

        // ---- layer 0: 3 chained WMMAs (K=96), fully unrolled ----
        layer_gates<K0>(zbuf, wT, bias, gbuf, mT, j, hi, ln, ascale, amul, aoff);
        barrier_lds();
        cell_update_stage(0, true, true, gbuf, hbuf, cbuf, zbuf, tid);
        barrier_lds();

        // ---- inner layers: single WMMA each (K=32) ----
        for (int l = 1; l < L_LAY; ++l) {
            const __bf16* wl = wT + WT0_ELEMS + (l - 1) * WTI_ELEMS;
            layer_gates<KI>(zbuf, wl, bias + l * NG, gbuf, mT, j, hi, ln,
                            ascale, amul, aoff);
            barrier_lds();
            cell_update_stage(l, l + 1 < L_LAY, false, gbuf, hbuf, cbuf, zbuf, tid);
            barrier_lds();
        }
    }

    // ================= outputs =================
    for (int idx = tid; idx < L_LAY * BM * H_HID; idx += NT) {
        int l  = idx / (BM * H_HID);
        int r  = idx % (BM * H_HID);
        int bl = r >> 3, h = r & 7;
        int bg = bBase + bl;
        if (bg < B_TOT)
            out[B_TOT + ((size_t)l * B_TOT + bg) * H_HID + h] = hbuf[idx];
    }
    if (tid < BM) {
        int bg = bBase + tid;
        if (bg < B_TOT) {
            const float* hf = hbuf + ((L_LAY - 1) * BM + tid) * H_HID;
            float s = by[0];
#pragma unroll
            for (int h = 0; h < H_HID; ++h) s += hf[h] * Why[h];
            out[bg] = s;
        }
    }
}

extern "C" void kernel_launch(void* const* d_in, const int* in_sizes, int n_in,
                              void* d_out, int out_size, void* d_ws, size_t ws_size,
                              hipStream_t stream) {
    (void)in_sizes; (void)n_in; (void)out_size; (void)d_ws; (void)ws_size;

    const int*   x   = (const int*)  d_in[0];
    const float* h0  = (const float*)d_in[1];
    const float* c0  = (const float*)d_in[2];
    const float* emb = (const float*)d_in[3];
    const float* Wx0 = (const float*)d_in[4];
    const float* Wh0 = (const float*)d_in[5];
    const float* b0  = (const float*)d_in[6];
    const float* Wx  = (const float*)d_in[7];
    const float* Wh  = (const float*)d_in[8];
    const float* bb  = (const float*)d_in[9];
    const float* Why = (const float*)d_in[10];
    const float* by  = (const float*)d_in[11];
    float* out = (float*)d_out;

    (void)hipFuncSetAttribute((const void*)lstm_wmma_kernel,
                              hipFuncAttributeMaxDynamicSharedMemorySize,
                              (int)SMEM_BYTES);

    const int grid = (B_TOT + BM - 1) / BM;   // 16 workgroups
    lstm_wmma_kernel<<<grid, NT, SMEM_BYTES, stream>>>(
        x, h0, c0, emb, Wx0, Wh0, b0, Wx, Wh, bb, Why, by, out);
}
